// ViSNetEncoderMSE_90477781057666
// MI455X (gfx1250) — matrix operand order repaired and checked by reading, not compiled, not run on hardware
//
#include <hip/hip_runtime.h>
#include <hip/hip_bf16.h>
#include <cstdint>

// ---------------- problem constants ----------------
#define N_NODES   10000
#define N_EDGES   160000
#define HIDDEN    128
#define NUM_RBF   32
#define NUM_LAYERS 2
#define LATENT    64
#define N_GRAPHS  128
#define OUTCOLS   (LATENT + 1)   // 65
#define OUTPAD    80             // 5 N-tiles of 16

typedef __bf16 bf16_t;
typedef __attribute__((ext_vector_type(16))) __bf16 bfx16;
typedef __attribute__((ext_vector_type(8)))  __bf16 bfx8;
typedef __attribute__((ext_vector_type(8)))  float  fx8;

#define WMMA_BF16(A, B, C) \
  __builtin_amdgcn_wmma_f32_16x16x32_bf16(false, (A), false, (B), (short)0, (C), false, false)

__device__ __forceinline__ int lane_id() { return (int)(threadIdx.x & 31u); }

__device__ __forceinline__ void cvt4(bfx16& a, int o, float4 q) {
  a[o + 0] = (bf16_t)q.x; a[o + 1] = (bf16_t)q.y;
  a[o + 2] = (bf16_t)q.z; a[o + 3] = (bf16_t)q.w;
}
__device__ __forceinline__ void mul4(bfx16& a, int o, float4 x, float4 f) {
  a[o + 0] = (bf16_t)(x.x * f.x); a[o + 1] = (bf16_t)(x.y * f.y);
  a[o + 2] = (bf16_t)(x.z * f.z); a[o + 3] = (bf16_t)(x.w * f.w);
}

// ---- A fragment: 16 rows x 32 K-slice from f32 row-major (vectorized) ----
// lanes 0-15: row=lane, K = k0+{0..7} then k0+{16..23}
// lanes 16-31: row=lane-16, K = k0+{8..15} then k0+{24..31}
__device__ __forceinline__ bfx16 load_a_f32(const float* base, int stride, int k0) {
  int l = lane_id();
  const float* r = base + (size_t)(l & 15) * stride + k0 + ((l >> 4) << 3);
  const float4* q = (const float4*)r;
  float4 q0 = q[0], q1 = q[1], q2 = q[4], q3 = q[5];
  bfx16 a;
  cvt4(a, 0, q0); cvt4(a, 4, q1); cvt4(a, 8, q2); cvt4(a, 12, q3);
  return a;
}

// A fragment from bf16 row-major (rbf): two 16B vector loads
__device__ __forceinline__ bfx16 load_a_bf16v(const bf16_t* base, int stride, int k0) {
  int l = lane_id();
  const bf16_t* r = base + (size_t)(l & 15) * stride + k0 + ((l >> 4) << 3);
  bfx8 lo = *(const bfx8*)r;
  bfx8 hi = *(const bfx8*)(r + 16);
  bfx16 a;
#pragma unroll
  for (int i = 0; i < 8; ++i) { a[i] = lo[i]; a[8 + i] = hi[i]; }
  return a;
}

// ---- B fragment from PACKED (fragment-major) weights: one 32B load ----
// packed layout: frag blocks of 512 bf16; lane l owns bf16[frag*512 + l*16 .. +15]
__device__ __forceinline__ bfx16 load_b_packed(const bf16_t* wp, int frag) {
  return ((const bfx16*)(wp + ((size_t)frag << 9)))[lane_id()];
}

// ---- C fragment load from f32 row-major ----
__device__ __forceinline__ fx8 load_c_f32(const float* base, int stride, int n0) {
  int l = lane_id();
  const float* p = base + (size_t)((l >> 4) << 3) * stride + n0 + (l & 15);
  fx8 c;
#pragma unroll
  for (int r = 0; r < 8; ++r) c[r] = p[(size_t)r * stride];
  return c;
}

__device__ __forceinline__ float silu_f(float v) { return v / (1.0f + __expf(-v)); }

// ======================= utility kernels =======================
// Pack f32 weights [K][N] -> bf16 fragment-major, N padded to Npad (zero fill).
// Fragment (kt,nt): lane l, elem i -> K = kt*32 + (l>>4)*8 + (i&7) + (i>>3)*16, N = nt*16 + (l&15)
__global__ void k_pack_w(const float* __restrict__ src, bf16_t* __restrict__ dst,
                         int K, int N, int Npad) {
  int idx = blockIdx.x * blockDim.x + threadIdx.x;
  if (idx >= K * Npad) return;
  int frag = idx >> 9;
  int rem  = idx & 511;
  int lane = rem >> 4;
  int i    = rem & 15;
  int NT = Npad >> 4;
  int kt = frag / NT, nt = frag - kt * NT;
  int k = kt * 32 + ((lane >> 4) << 3) + (i & 7) + ((i >> 3) << 4);
  int n = nt * 16 + (lane & 15);
  dst[idx] = (n < N) ? (bf16_t)src[(size_t)k * N + n] : (bf16_t)0.0f;
}

__global__ void k_zero_f32v(float4* __restrict__ p, int n4) {
  int i = blockIdx.x * blockDim.x + threadIdx.x;
  float4 z = {0.f, 0.f, 0.f, 0.f};
  for (; i < n4; i += gridDim.x * blockDim.x) p[i] = z;
}

__global__ void k_vec_addv(float4* __restrict__ dst, const float4* __restrict__ src, int n4) {
  int i = blockIdx.x * blockDim.x + threadIdx.x;
  for (; i < n4; i += gridDim.x * blockDim.x) {
    float4 a = dst[i], b = src[i];
    a.x += b.x; a.y += b.y; a.z += b.z; a.w += b.w;
    dst[i] = a;
  }
}

// x[n][h] = embed_table[z[n]][h]  (float4 rows)
__global__ void k_embed(const int* __restrict__ z, const float4* __restrict__ tab,
                        float4* __restrict__ x) {
  int i = blockIdx.x * blockDim.x + threadIdx.x;
  if (i >= N_NODES * (HIDDEN / 4)) return;
  int n = i >> 5, c = i & 31;
  x[i] = tab[(size_t)z[n] * 32 + c];
}

// edge geometry + expnorm RBF (bf16, vector stores)
__global__ void k_edge_geom(const int* __restrict__ ei, const float* __restrict__ pos,
                            float* __restrict__ dirn, bf16_t* __restrict__ rbf) {
  int e = blockIdx.x * blockDim.x + threadIdx.x;
  if (e >= N_EDGES) return;
  int s = ei[e], t = ei[N_EDGES + e];
  float dx = pos[t * 3 + 0] - pos[s * 3 + 0];
  float dy = pos[t * 3 + 1] - pos[s * 3 + 1];
  float dz = pos[t * 3 + 2] - pos[s * 3 + 2];
  float r = sqrtf(dx * dx + dy * dy + dz * dz + 1e-8f);
  float inv = 1.0f / r;
  dirn[e * 3 + 0] = dx * inv;
  dirn[e * 3 + 1] = dy * inv;
  dirn[e * 3 + 2] = dz * inv;
  const float m0 = 0.006737947f;             // exp(-5)
  const float step = (1.0f - m0) / 31.0f;
  const float w = (2.0f / 32.0f) * (1.0f - m0);
  const float beta = 1.0f / (w * w);
  float cut = (r < 5.0f) ? 0.5f * (__cosf(3.14159265358979f * r * 0.2f) + 1.0f) : 0.0f;
  float er = __expf(-r);
  bfx8 v0, v1, v2, v3;
#pragma unroll
  for (int k = 0; k < 8; ++k) {
    float d0 = er - (m0 + (float)k * step);
    float d1 = er - (m0 + (float)(k + 8) * step);
    float d2 = er - (m0 + (float)(k + 16) * step);
    float d3 = er - (m0 + (float)(k + 24) * step);
    v0[k] = (bf16_t)(cut * __expf(-beta * d0 * d0));
    v1[k] = (bf16_t)(cut * __expf(-beta * d1 * d1));
    v2[k] = (bf16_t)(cut * __expf(-beta * d2 * d2));
    v3[k] = (bf16_t)(cut * __expf(-beta * d3 * d3));
  }
  bfx8* out = (bfx8*)(rbf + (size_t)e * NUM_RBF);
  out[0] = v0; out[1] = v1; out[2] = v2; out[3] = v3;
}

// ======================= WMMA GEMM kernels =======================
// D[r][:] = A[r][0:128] @ W(packed 128x128)
__global__ void __launch_bounds__(32) k_gemm_rows128(const float* __restrict__ A,
                                                     const bf16_t* __restrict__ Wp,
                                                     float* __restrict__ D) {
  int row0 = blockIdx.x * 16;
  bfx16 a[4];
#pragma unroll
  for (int ks = 0; ks < 4; ++ks) a[ks] = load_a_f32(A + (size_t)row0 * 128, 128, ks * 32);
  int l = lane_id();
  int rb = (l >> 4) << 3;
#pragma unroll
  for (int nt = 0; nt < 8; ++nt) {
    fx8 acc = {};
#pragma unroll
    for (int ks = 0; ks < 4; ++ks)
      acc = WMMA_BF16(a[ks], load_b_packed(Wp, ks * 8 + nt), acc);
    int col = nt * 16 + (l & 15);
#pragma unroll
    for (int r = 0; r < 8; ++r) D[(size_t)(row0 + rb + r) * 128 + col] = acc[r];
  }
}

// x[tile] = x[tile] + dx[tile] @ Wo(packed)
__global__ void __launch_bounds__(32) k_x_update(float* __restrict__ x,
                                                 const float* __restrict__ dxb,
                                                 const bf16_t* __restrict__ Wp) {
  int row0 = blockIdx.x * 16;
  bfx16 a[4];
#pragma unroll
  for (int ks = 0; ks < 4; ++ks) a[ks] = load_a_f32(dxb + (size_t)row0 * 128, 128, ks * 32);
  int l = lane_id();
  int rb = (l >> 4) << 3;
#pragma unroll
  for (int nt = 0; nt < 8; ++nt) {
    fx8 acc = load_c_f32(x + (size_t)row0 * 128, 128, nt * 16);
#pragma unroll
    for (int ks = 0; ks < 4; ++ks)
      acc = WMMA_BF16(a[ks], load_b_packed(Wp, ks * 8 + nt), acc);
    int col = nt * 16 + (l & 15);
#pragma unroll
    for (int r = 0; r < 8; ++r) x[(size_t)(row0 + rb + r) * 128 + col] = acc[r];
  }
}

// ======================= fused edge message kernel =======================
__global__ void __launch_bounds__(32) k_edge_msg(const int* __restrict__ ei,
                                                 const float* __restrict__ x,
                                                 const float* __restrict__ vmix,
                                                 const float* __restrict__ dirn,
                                                 const bf16_t* __restrict__ rbf,
                                                 const bf16_t* __restrict__ Wrbf_p,
                                                 const bf16_t* __restrict__ W1_p,
                                                 float* __restrict__ dxb,
                                                 float* __restrict__ vdelta) {
  __shared__ __align__(16) float xs_s[16][128];    // staged x[src] rows (async)
  __shared__ __align__(16) float filt_s[16][128];
  __shared__ __align__(16) float phi_s[16][128];
  __shared__ int   src_s[16];
  __shared__ int   dst_s[16];
  __shared__ float dir_s[16][3];

  int e0 = blockIdx.x * 16;
  int l = lane_id();
  if (l < 16) {
    src_s[l] = ei[e0 + l];
    dst_s[l] = ei[N_EDGES + e0 + l];
    dir_s[l][0] = dirn[(size_t)(e0 + l) * 3 + 0];
    dir_s[l][1] = dirn[(size_t)(e0 + l) * 3 + 1];
    dir_s[l][2] = dirn[(size_t)(e0 + l) * 3 + 2];
  }
  __syncthreads();

  // --- async stage: 16 x[src] rows -> LDS (overlapped with filt WMMA) ---
  // 16 rows * 128 f32 = 512 x 16B chunks; 16 per lane
#pragma unroll
  for (int it = 0; it < 16; ++it) {
    int c = it * 32 + l;
    int row = c >> 5, chunk = c & 31;
    const float* g = x + (size_t)src_s[row] * 128 + chunk * 4;
    unsigned loff = (unsigned)(uintptr_t)(&xs_s[row][chunk * 4]);
    asm volatile("global_load_async_to_lds_b128 %0, %1, off"
                 :: "v"(loff), "v"(g) : "memory");
  }

  int rb = (l >> 4) << 3;
  int nlo = l & 15;

  // filt = rbf(16x32) @ W_rbf(32x128): one WMMA per 16-col tile
  bfx16 ar = load_a_bf16v(rbf + (size_t)e0 * NUM_RBF, NUM_RBF, 0);
#pragma unroll
  for (int nt = 0; nt < 8; ++nt) {
    fx8 acc = {};
    acc = WMMA_BF16(ar, load_b_packed(Wrbf_p, nt), acc);
    int col = nt * 16 + nlo;
#pragma unroll
    for (int r = 0; r < 8; ++r) filt_s[rb + r][col] = acc[r];
  }

  asm volatile("s_wait_asynccnt 0" ::: "memory");
  __syncthreads();

  // A fragments for t = x[src] * filt (bf16), K = 128 (from LDS, vectorized)
  bfx16 at[4];
#pragma unroll
  for (int ks = 0; ks < 4; ++ks) {
    int m = l & 15;
    int kb = ks * 32 + ((l >> 4) << 3);
    const float4* xq = (const float4*)&xs_s[m][kb];
    const float4* fq = (const float4*)&filt_s[m][kb];
    float4 x0 = xq[0], x1 = xq[1], x2 = xq[4], x3 = xq[5];
    float4 f0 = fq[0], f1 = fq[1], f2 = fq[4], f3 = fq[5];
    mul4(at[ks], 0, x0, f0); mul4(at[ks], 4, x1, f1);
    mul4(at[ks], 8, x2, f2); mul4(at[ks], 12, x3, f3);
  }

  // phi = silu(t @ W1); scatter dx[dst] += phi
#pragma unroll
  for (int nt = 0; nt < 8; ++nt) {
    fx8 acc = {};
#pragma unroll
    for (int ks = 0; ks < 4; ++ks)
      acc = WMMA_BF16(at[ks], load_b_packed(W1_p, ks * 8 + nt), acc);
    int col = nt * 16 + nlo;
#pragma unroll
    for (int r = 0; r < 8; ++r) {
      float v = silu_f(acc[r]);
      phi_s[rb + r][col] = v;
      atomicAdd(&dxb[(size_t)dst_s[rb + r] * 128 + col], v);
    }
  }
  __syncthreads();

  // vmsg scatter: vdelta[dst][c][h] += vmix[src][c][h]*filt + phi*dirn[c]
  for (int idx = l; idx < 16 * 128; idx += 32) {
    int row = idx >> 7, col = idx & 127;
    float f = filt_s[row][col];
    float p = phi_s[row][col];
    const float* vm = vmix + (size_t)src_s[row] * 384 + col;
    float* vd = vdelta + (size_t)dst_s[row] * 384 + col;
#pragma unroll
    for (int c = 0; c < 3; ++c)
      atomicAdd(&vd[c * 128], vm[c * 128] * f + p * dir_s[row][c]);
  }
}

// ======================= head kernel =======================
__global__ void __launch_bounds__(32) k_head(const float* __restrict__ x,
                                             const float* __restrict__ v2,
                                             const int* __restrict__ batch,
                                             const bf16_t* __restrict__ Wa_p,
                                             const float* __restrict__ ba,
                                             const bf16_t* __restrict__ Wb_p,
                                             const float* __restrict__ bb,
                                             float* __restrict__ pooled) {
  __shared__ __align__(16) float vn_s[16][128];
  __shared__ __align__(16) float a_s[16][128];
  int n0 = blockIdx.x * 16;
  int l = lane_id();
  int rb = (l >> 4) << 3;
  int nlo = l & 15;

  for (int idx = l; idx < 16 * 128; idx += 32) {
    int row = idx >> 7, col = idx & 127;
    const float* vr = v2 + (size_t)(n0 + row) * 384 + col;
    float a0 = vr[0], a1 = vr[128], a2 = vr[256];
    vn_s[row][col] = sqrtf(a0 * a0 + a1 * a1 + a2 * a2 + 1e-8f);
  }
  __syncthreads();

  // a = silu([x, vnorm] @ W_a + b_a), K=256
  bfx16 af[8];
#pragma unroll
  for (int ks = 0; ks < 4; ++ks) af[ks] = load_a_f32(x + (size_t)n0 * 128, 128, ks * 32);
#pragma unroll
  for (int ks = 0; ks < 4; ++ks) af[4 + ks] = load_a_f32(&vn_s[0][0], 128, ks * 32);

#pragma unroll
  for (int nt = 0; nt < 8; ++nt) {
    fx8 acc = {};
#pragma unroll
    for (int ks = 0; ks < 8; ++ks)
      acc = WMMA_BF16(af[ks], load_b_packed(Wa_p, ks * 8 + nt), acc);
    int col = nt * 16 + nlo;
    float bav = ba[col];
#pragma unroll
    for (int r = 0; r < 8; ++r) a_s[rb + r][col] = silu_f(acc[r] + bav);
  }
  __syncthreads();

  // out = a @ W_b (+ b_b), pool by graph (W_b packed to 80 cols, zero-padded)
  bfx16 ao[4];
#pragma unroll
  for (int ks = 0; ks < 4; ++ks) ao[ks] = load_a_f32(&a_s[0][0], 128, ks * 32);
#pragma unroll
  for (int nt = 0; nt < 5; ++nt) {
    fx8 acc = {};
#pragma unroll
    for (int ks = 0; ks < 4; ++ks)
      acc = WMMA_BF16(ao[ks], load_b_packed(Wb_p, ks * 5 + nt), acc);
    int col = nt * 16 + nlo;
    if (col < OUTCOLS) {
      float bbv = bb[col];
#pragma unroll
      for (int r = 0; r < 8; ++r) {
        int node = n0 + rb + r;
        atomicAdd(&pooled[(size_t)batch[node] * OUTCOLS + col], acc[r] + bbv);
      }
    }
  }
}

// pooled -> (mu, log_var) flat
__global__ void k_writeout(const float* __restrict__ pooled, float* __restrict__ out) {
  int i = blockIdx.x * blockDim.x + threadIdx.x;
  if (i < N_GRAPHS * LATENT) {
    int g = i >> 6, j = i & 63;
    out[i] = pooled[g * OUTCOLS + j];
  } else if (i < N_GRAPHS * LATENT + N_GRAPHS) {
    int g = i - N_GRAPHS * LATENT;
    float v = pooled[g * OUTCOLS + LATENT];
    out[N_GRAPHS * LATENT + g] = fminf(fmaxf(v, -10.0f), 2.0f);
  }
}

// ======================= host side =======================
extern "C" void kernel_launch(void* const* d_in, const int* in_sizes, int n_in,
                              void* d_out, int out_size, void* d_ws, size_t ws_size,
                              hipStream_t stream) {
  const int*   z      = (const int*)d_in[0];
  const float* pos    = (const float*)d_in[1];
  const int*   batch  = (const int*)d_in[2];
  const int*   ei     = (const int*)d_in[3];
  const float* emb    = (const float*)d_in[4];
  const float* Wrbf   = (const float*)d_in[5];
  const float* W1     = (const float*)d_in[6];
  const float* Wo     = (const float*)d_in[7];
  const float* Wv     = (const float*)d_in[8];
  // d_in[9] = Wvec1 (unused by the reference)
  const float* Wvec2  = (const float*)d_in[10];
  const float* Wa     = (const float*)d_in[11];
  const float* ba     = (const float*)d_in[12];
  const float* Wb     = (const float*)d_in[13];
  const float* bb     = (const float*)d_in[14];
  float* out = (float*)d_out;

  // ---- carve workspace ----
  char* cur = (char*)d_ws;
  auto alloc = [&](size_t bytes) -> char* {
    char* p = cur;
    cur += (bytes + 255) & ~(size_t)255;
    return p;
  };
  bf16_t* wrbf_p  = (bf16_t*)alloc((size_t)NUM_LAYERS * NUM_RBF * HIDDEN * 2);
  bf16_t* w1_p    = (bf16_t*)alloc((size_t)NUM_LAYERS * HIDDEN * HIDDEN * 2);
  bf16_t* wo_p    = (bf16_t*)alloc((size_t)NUM_LAYERS * HIDDEN * HIDDEN * 2);
  bf16_t* wv_p    = (bf16_t*)alloc((size_t)NUM_LAYERS * HIDDEN * HIDDEN * 2);
  bf16_t* wvec2_p = (bf16_t*)alloc((size_t)HIDDEN * HIDDEN * 2);
  bf16_t* wa_p    = (bf16_t*)alloc((size_t)2 * HIDDEN * HIDDEN * 2);
  bf16_t* wb_p    = (bf16_t*)alloc((size_t)HIDDEN * OUTPAD * 2);
  bf16_t* rbf_b   = (bf16_t*)alloc((size_t)N_EDGES * NUM_RBF * 2);
  float*  dirn    = (float*)alloc((size_t)N_EDGES * 3 * 4);
  float*  xbuf    = (float*)alloc((size_t)N_NODES * HIDDEN * 4);
  float*  dxbuf   = (float*)alloc((size_t)N_NODES * HIDDEN * 4);
  float*  vec     = (float*)alloc((size_t)N_NODES * 3 * HIDDEN * 4);
  float*  vdelta  = (float*)alloc((size_t)N_NODES * 3 * HIDDEN * 4);
  float*  vmix    = (float*)alloc((size_t)N_NODES * 3 * HIDDEN * 4);   // reused as v2
  float*  pooled  = (float*)alloc((size_t)N_GRAPHS * OUTCOLS * 4);

  const int T = 256;
  auto cdiv = [](int a, int b) { return (a + b - 1) / b; };

  // pack weights into fragment-major bf16
  for (int l = 0; l < NUM_LAYERS; ++l) {
    k_pack_w<<<cdiv(NUM_RBF * HIDDEN, T), T, 0, stream>>>(
        Wrbf + (size_t)l * NUM_RBF * HIDDEN, wrbf_p + (size_t)l * NUM_RBF * HIDDEN,
        NUM_RBF, HIDDEN, HIDDEN);
    k_pack_w<<<cdiv(HIDDEN * HIDDEN, T), T, 0, stream>>>(
        W1 + (size_t)l * HIDDEN * HIDDEN, w1_p + (size_t)l * HIDDEN * HIDDEN,
        HIDDEN, HIDDEN, HIDDEN);
    k_pack_w<<<cdiv(HIDDEN * HIDDEN, T), T, 0, stream>>>(
        Wo + (size_t)l * HIDDEN * HIDDEN, wo_p + (size_t)l * HIDDEN * HIDDEN,
        HIDDEN, HIDDEN, HIDDEN);
    k_pack_w<<<cdiv(HIDDEN * HIDDEN, T), T, 0, stream>>>(
        Wv + (size_t)l * HIDDEN * HIDDEN, wv_p + (size_t)l * HIDDEN * HIDDEN,
        HIDDEN, HIDDEN, HIDDEN);
  }
  k_pack_w<<<cdiv(HIDDEN * HIDDEN, T), T, 0, stream>>>(Wvec2, wvec2_p, HIDDEN, HIDDEN, HIDDEN);
  k_pack_w<<<cdiv(2 * HIDDEN * HIDDEN, T), T, 0, stream>>>(Wa, wa_p, 2 * HIDDEN, HIDDEN, HIDDEN);
  k_pack_w<<<cdiv(HIDDEN * OUTPAD, T), T, 0, stream>>>(Wb, wb_p, HIDDEN, OUTCOLS, OUTPAD);

  // init
  k_zero_f32v<<<1024, T, 0, stream>>>((float4*)vec, N_NODES * 3 * HIDDEN / 4);
  k_zero_f32v<<<16, T, 0, stream>>>((float4*)pooled, N_GRAPHS * OUTCOLS / 4 + 1);
  k_edge_geom<<<cdiv(N_EDGES, T), T, 0, stream>>>(ei, pos, dirn, rbf_b);
  k_embed<<<cdiv(N_NODES * HIDDEN / 4, T), T, 0, stream>>>(z, (const float4*)emb, (float4*)xbuf);

  for (int l = 0; l < NUM_LAYERS; ++l) {
    k_zero_f32v<<<1024, T, 0, stream>>>((float4*)dxbuf, N_NODES * HIDDEN / 4);
    k_zero_f32v<<<1024, T, 0, stream>>>((float4*)vdelta, N_NODES * 3 * HIDDEN / 4);
    // vmix = vec @ Wv[l]  (30000 x 128 rows)
    k_gemm_rows128<<<(N_NODES * 3) / 16, 32, 0, stream>>>(vec, wv_p + (size_t)l * HIDDEN * HIDDEN, vmix);
    // fused edge messages
    k_edge_msg<<<N_EDGES / 16, 32, 0, stream>>>(ei, xbuf, vmix, dirn, rbf_b,
                                                wrbf_p + (size_t)l * NUM_RBF * HIDDEN,
                                                w1_p + (size_t)l * HIDDEN * HIDDEN,
                                                dxbuf, vdelta);
    // x += dx @ Wo[l]
    k_x_update<<<N_NODES / 16, 32, 0, stream>>>(xbuf, dxbuf, wo_p + (size_t)l * HIDDEN * HIDDEN);
    // vec += vdelta
    k_vec_addv<<<1024, T, 0, stream>>>((float4*)vec, (const float4*)vdelta, N_NODES * 3 * HIDDEN / 4);
  }

  // v2 = vec @ Wvec2 (reuse vmix buffer)
  k_gemm_rows128<<<(N_NODES * 3) / 16, 32, 0, stream>>>(vec, wvec2_p, vmix);
  // head: vnorm, silu MLP, output projection, pooling
  k_head<<<N_NODES / 16, 32, 0, stream>>>(xbuf, vmix, batch, wa_p, ba, wb_p, bb, pooled);
  // final write
  k_writeout<<<cdiv(N_GRAPHS * LATENT + N_GRAPHS, T), T, 0, stream>>>(pooled, out);
}
